// CrystalGraphConvNet_12189117186415
// MI455X (gfx1250) — matrix operand order repaired and checked by reading, not compiled, not run on hardware
//
#include <hip/hip_runtime.h>
#include <hip/hip_bf16.h>

typedef __bf16 bf16;
typedef _Float16 f16;
typedef __attribute__((ext_vector_type(16))) __bf16 v16bf;
typedef __attribute__((ext_vector_type(8)))  float  v8f;

#define N_ATOMS 100000
#define N_EDGES 1200000
#define N_CRYS  3125
#define ORIG 92
#define NBRF 41
#define AFEA 64
#define NCONV 3
#define HFEA 128

__device__ __forceinline__ float softplus_f(float x) {
  return (x > 20.f) ? x : log1pf(__expf(x));
}
__device__ __forceinline__ float sigmoid_f(float x) {
  return 1.f / (1.f + __expf(-x));
}

// ---------------------------------------------------------------------------
// Convert atom_fea (N x 92 f32) -> bf16 padded to (N x 96)
// ---------------------------------------------------------------------------
__global__ void prep_atoms_kernel(const float* __restrict__ a, bf16* __restrict__ o) {
  const int i = blockIdx.x * 256 + threadIdx.x;
  if (i >= N_ATOMS * 96) return;
  const int n = i / 96, j = i % 96;
  o[i] = (j < ORIG) ? (bf16)a[(size_t)n * ORIG + j] : (bf16)0.0f;
}

// ---------------------------------------------------------------------------
// Pre-swizzle W (Ksrc x Ncols, f32, row-major) into per-lane WMMA B fragments.
// B layout (16-bit, 32x16 per k-step): lane l holds column N = nt*16 + (l&15),
// dword p holds K pair (kt*32 + 2p + (l>=16 ? 8 : 0) + (p>=4 ? 8 : 0)).
// dst layout: [(kt*nnt + nt)*32 + lane] -> v16bf (32 bytes)
// ---------------------------------------------------------------------------
__global__ void build_wfrag_kernel(const float* __restrict__ W, v16bf* __restrict__ dst,
                                   int Ksrc, int Ncols, int nkt, int nnt) {
  const int tid = blockIdx.x * 64 + threadIdx.x;
  if (tid >= nkt * nnt * 32) return;
  const int lane = tid & 31;
  const int tileid = tid >> 5;
  const int nt = tileid % nnt, kt = tileid / nnt;
  const int n = nt * 16 + (lane & 15);
  const int koff = (lane >= 16) ? 8 : 0;
  union { v16bf v; bf16 h[16]; } f;
#pragma unroll
  for (int p = 0; p < 8; ++p) {
    const int K = kt * 32 + 2 * p + koff + ((p >= 4) ? 8 : 0);
    f.h[2 * p]     = (K     < Ksrc) ? (bf16)W[(size_t)K * Ncols + n]       : (bf16)0.0f;
    f.h[2 * p + 1] = (K + 1 < Ksrc) ? (bf16)W[(size_t)(K + 1) * Ncols + n] : (bf16)0.0f;
  }
  dst[tid] = f.v;
}

// ---------------------------------------------------------------------------
// Embedding GEMM: x = atom_bf16 (N x 96) @ emb_W (96 x 64) + b, bf16 WMMA,
// 32 rows per block, 8 waves = 2 row-strips x 4 N-tiles.
// ---------------------------------------------------------------------------
__global__ __launch_bounds__(256) void emb_gemm_kernel(
    const bf16* __restrict__ abf, const v16bf* __restrict__ wfrag,
    const float* __restrict__ bias, float* __restrict__ xout, bf16* __restrict__ xbfout) {
  __shared__ bf16 tile[32][96];
  const int t = threadIdx.x;
  const int r0 = blockIdx.x * 32;
  {
    const int r = t >> 3, q = t & 7;  // 8 threads/row, 96 bf16 = 24 x u64
    const unsigned long long* src = (const unsigned long long*)(abf + (size_t)(r0 + r) * 96);
    unsigned long long* dst = (unsigned long long*)(&tile[r][0]);
    dst[q] = src[q]; dst[q + 8] = src[q + 8]; dst[q + 16] = src[q + 16];
  }
  __syncthreads();
  const int wave = t >> 5, lane = t & 31;
  const int strip = wave >> 2, nt = wave & 3;
  const int nloc = lane & 15;
  const int ngl = nt * 16 + nloc;
  const int hi = lane >> 4;
  const int kb4 = hi * 4;
  v8f acc = {0.f, 0.f, 0.f, 0.f, 0.f, 0.f, 0.f, 0.f};
  const unsigned* rowp = (const unsigned*)(&tile[strip * 16 + nloc][0]);
#pragma unroll
  for (int kt = 0; kt < 3; ++kt) {
    union { v16bf v; unsigned u[8]; } af;
    const int kb = kt * 16 + kb4;
#pragma unroll
    for (int p = 0; p < 8; ++p) af.u[p] = rowp[kb + p + ((p >= 4) ? 4 : 0)];
    const v16bf bf = wfrag[(kt * 4 + nt) * 32 + lane];
    acc = __builtin_amdgcn_wmma_f32_16x16x32_bf16(false, af.v, false, bf,
                                                  (short)0, acc, false, false);
  }
  const float bval = bias[ngl];
  const int row0 = r0 + strip * 16 + hi * 8;
#pragma unroll
  for (int r8 = 0; r8 < 8; ++r8) {
    const float v = acc[r8] + bval;
    xout[(size_t)(row0 + r8) * AFEA + ngl] = v;
    xbfout[(size_t)(row0 + r8) * AFEA + ngl] = (bf16)v;
  }
}

// ---------------------------------------------------------------------------
// Conv GEMM: C(E x 128) = A(E x 192pad) @ W(192 x 128) + b, A gathered on the
// fly: [x[self] | x[nbr] | bf16(nbr_fea) | 0pad]. 64 edges/block, 8 waves each
// own one 16-column N-tile, 6 K-steps. Fused BN column sum/sumsq partials
// (exact f32 from accumulators); C stored as f16 to halve HBM traffic.
// ---------------------------------------------------------------------------
__global__ __launch_bounds__(256) void conv_gemm_kernel(
    const bf16* __restrict__ xbf, const float* __restrict__ nbr_fea,
    const int* __restrict__ self_idx, const int* __restrict__ nbr_idx,
    const v16bf* __restrict__ wfrag, const float* __restrict__ bias,
    f16* __restrict__ Cout, float* __restrict__ sumbuf, float* __restrict__ sqbuf) {
  __shared__ bf16 tile[64][192];
  const int t = threadIdx.x;
  const int e0 = blockIdx.x * 64;
  {
    const int r = t >> 2, q = t & 3;  // 4 threads/row
    const int e = e0 + r;
    const int si = self_idx[e];
    const int ni = nbr_idx[e];
    const uint4* srcs = (const uint4*)(xbf + (size_t)si * AFEA);
    const uint4* srcn = (const uint4*)(xbf + (size_t)ni * AFEA);
    uint4* ds = (uint4*)(&tile[r][0]);
    uint4* dn = (uint4*)(&tile[r][AFEA]);
    ds[q * 2]     = srcs[q * 2];
    ds[q * 2 + 1] = srcs[q * 2 + 1];
    dn[q * 2]     = srcn[q * 2];
    dn[q * 2 + 1] = srcn[q * 2 + 1];
    const float* nf = nbr_fea + (size_t)e * NBRF;
    for (int i = q; i < 64; i += 4)
      tile[r][128 + i] = (i < NBRF) ? (bf16)nf[i] : (bf16)0.0f;
  }
  __syncthreads();

  const int wave = t >> 5, lane = t & 31;
  const int nloc = lane & 15;
  const int ngl = wave * 16 + nloc;
  const int hi = lane >> 4;
  const int kb4 = hi * 4;
  const int mbase = hi * 8;

  v16bf bfr[6];
#pragma unroll
  for (int kt = 0; kt < 6; ++kt) bfr[kt] = wfrag[(kt * 8 + wave) * 32 + lane];
  const float bval = bias[ngl];

  float scol = 0.f, qcol = 0.f;
  for (int strip = 0; strip < 4; ++strip) {
    v8f acc = {0.f, 0.f, 0.f, 0.f, 0.f, 0.f, 0.f, 0.f};
    const unsigned* rowp = (const unsigned*)(&tile[strip * 16 + nloc][0]);
#pragma unroll
    for (int kt = 0; kt < 6; ++kt) {
      union { v16bf v; unsigned u[8]; } af;
      const int kb = kt * 16 + kb4;
#pragma unroll
      for (int p = 0; p < 8; ++p) af.u[p] = rowp[kb + p + ((p >= 4) ? 4 : 0)];
      acc = __builtin_amdgcn_wmma_f32_16x16x32_bf16(false, af.v, false, bfr[kt],
                                                    (short)0, acc, false, false);
    }
    f16* cp = Cout + (size_t)(e0 + strip * 16 + mbase) * 128 + ngl;
#pragma unroll
    for (int r8 = 0; r8 < 8; ++r8) {
      const float v = acc[r8] + bval;
      cp[(size_t)r8 * 128] = (f16)v;
      scol += v; qcol += v * v;
    }
  }
  // lanes l and l^16 hold the same output column -> combine, one atomic per col
  scol += __shfl_xor(scol, 16, 32);
  qcol += __shfl_xor(qcol, 16, 32);
  if (hi == 0) {
    const int slot = (blockIdx.x & 63) * 128;
    atomicAdd(&sumbuf[slot + ngl], scol);
    atomicAdd(&sqbuf[slot + ngl], qcol);
  }
}

// ---------------------------------------------------------------------------
// Fold BN partial sums into (scale, shift): y = x*scale + shift
// ---------------------------------------------------------------------------
__global__ void bn_finalize_kernel(const float* __restrict__ sumbuf, const float* __restrict__ sqbuf,
                                   int nslots, int C, float invcount,
                                   const float* __restrict__ g, const float* __restrict__ b,
                                   float* __restrict__ scale, float* __restrict__ shift) {
  const int c = threadIdx.x;
  if (c >= C) return;
  float s = 0.f, q = 0.f;
  for (int i = 0; i < nslots; ++i) { s += sumbuf[i * C + c]; q += sqbuf[i * C + c]; }
  const float m = s * invcount;
  const float var = q * invcount - m * m;
  const float inv = rsqrtf(var + 1e-5f);
  const float sc = g[c] * inv;
  scale[c] = sc;
  shift[c] = b[c] - m * sc;
}

// ---------------------------------------------------------------------------
// Pass 2: BN apply + sigmoid(filt)*softplus(core) + segment-sum.
// self_fea_idx is sorted: each 64-thread group walks 16 consecutive edges and
// run-length-compresses atomics (one atomicAdd per (run, column)).
// ---------------------------------------------------------------------------
__global__ __launch_bounds__(256) void conv_msg_kernel(
    const f16* __restrict__ C, const float* __restrict__ scale, const float* __restrict__ shift,
    const int* __restrict__ self_idx, float* __restrict__ summed) {
  const int g = threadIdx.x >> 6;
  const int c = threadIdx.x & 63;
  const int e0 = (blockIdx.x * 4 + g) * 16;
  const float sc_f = scale[c],      sh_f = shift[c];
  const float sc_c = scale[64 + c], sh_c = shift[64 + c];
  float acc = 0.f;
  int cur = self_idx[e0];
#pragma unroll 4
  for (int k = 0; k < 16; ++k) {
    const int e = e0 + k;
    const size_t base = (size_t)e * 128;
    const float f  = (float)C[base + c] * sc_f + sh_f;
    const float co = (float)C[base + 64 + c] * sc_c + sh_c;
    const float msg = sigmoid_f(f) * softplus_f(co);
    const int s = self_idx[e];
    if (s != cur) {  // uniform across the 64-thread group
      atomicAdd(&summed[(size_t)cur * AFEA + c], acc);
      cur = s;
      acc = 0.f;
    }
    acc += msg;
  }
  atomicAdd(&summed[(size_t)cur * AFEA + c], acc);
}

// ---------------------------------------------------------------------------
// Column stats over an (rows x 64) matrix
// ---------------------------------------------------------------------------
__global__ __launch_bounds__(256) void colstats64_kernel(const float* __restrict__ X, int rows,
                                                         float* __restrict__ sum, float* __restrict__ sq) {
  const int t = threadIdx.x;
  const int c = t & 63, g = t >> 6;
  float s = 0.f, q = 0.f;
  for (int r = blockIdx.x * 4 + g; r < rows; r += gridDim.x * 4) {
    const float v = X[(size_t)r * 64 + c];
    s += v; q += v * v;
  }
  __shared__ float ls[256], lq[256];
  ls[t] = s; lq[t] = q;
  __syncthreads();
  if (g == 0) {
    s = ls[c] + ls[64 + c] + ls[128 + c] + ls[192 + c];
    q = lq[c] + lq[64 + c] + lq[128 + c] + lq[192 + c];
    atomicAdd(&sum[c], s);
    atomicAdd(&sq[c], q);
  }
}

// ---------------------------------------------------------------------------
// x = softplus(x + bn2(summed));  also refresh bf16 copy for next conv gathers
// ---------------------------------------------------------------------------
__global__ void update_x_kernel(const float* __restrict__ summed,
                                const float* __restrict__ scale, const float* __restrict__ shift,
                                float* __restrict__ x, bf16* __restrict__ xbf) {
  const size_t i = (size_t)blockIdx.x * 256 + threadIdx.x;
  const int c = (int)(i & 63);
  float v = x[i] + summed[i] * scale[c] + shift[c];
  v = softplus_f(v);
  x[i] = v;
  xbf[i] = (bf16)v;
}

// ---------------------------------------------------------------------------
// Per-crystal sum + count
// ---------------------------------------------------------------------------
__global__ __launch_bounds__(256) void pool_kernel(const float* __restrict__ x,
                                                   const int* __restrict__ cidx,
                                                   float* __restrict__ csum,
                                                   float* __restrict__ counts) {
  const int a = blockIdx.x * 4 + (threadIdx.x >> 6);
  const int c = threadIdx.x & 63;
  const int k = cidx[a];
  atomicAdd(&csum[(size_t)k * AFEA + c], x[(size_t)a * AFEA + c]);
  if (c == 0) atomicAdd(&counts[k], 1.0f);
}

// ---------------------------------------------------------------------------
// Head: crys -> sp -> fc1(64x128) -> sp -> out(128x1). One block per crystal.
// ---------------------------------------------------------------------------
__global__ __launch_bounds__(128) void head_kernel(
    const float* __restrict__ csum, const float* __restrict__ counts,
    const float* __restrict__ fc1W, const float* __restrict__ fc1b,
    const float* __restrict__ outW, const float* __restrict__ outb,
    float* __restrict__ out) {
  const int k = blockIdx.x;
  const int t = threadIdx.x;
  __shared__ float sx[64];
  __shared__ float red[128];
  const float cnt = fmaxf(counts[k], 1.0f);
  if (t < 64) sx[t] = softplus_f(csum[(size_t)k * AFEA + t] / cnt);
  __syncthreads();
  float h = fc1b[t];
#pragma unroll 8
  for (int c = 0; c < 64; ++c) h += sx[c] * fc1W[c * HFEA + t];
  h = softplus_f(h);
  red[t] = h * outW[t];
  __syncthreads();
  for (int s = 64; s > 0; s >>= 1) {
    if (t < s) red[t] += red[t + s];
    __syncthreads();
  }
  if (t == 0) out[k] = red[0] + outb[0];
}

// ---------------------------------------------------------------------------
extern "C" void kernel_launch(void* const* d_in, const int* in_sizes, int n_in,
                              void* d_out, int out_size, void* d_ws, size_t ws_size,
                              hipStream_t stream) {
  const float* atom_fea = (const float*)d_in[0];
  const float* nbr_fea  = (const float*)d_in[1];
  const int*   self_idx = (const int*)d_in[2];
  const int*   nbr_idx  = (const int*)d_in[3];
  const int*   cry_idx  = (const int*)d_in[4];
  const float* emb_W    = (const float*)d_in[5];
  const float* emb_b    = (const float*)d_in[6];
  const float* conv_W   = (const float*)d_in[7];
  const float* conv_b   = (const float*)d_in[8];
  const float* bn1_g    = (const float*)d_in[9];
  const float* bn1_b    = (const float*)d_in[10];
  const float* bn2_g    = (const float*)d_in[11];
  const float* bn2_b    = (const float*)d_in[12];
  const float* fc1_W    = (const float*)d_in[13];
  const float* fc1_b    = (const float*)d_in[14];
  const float* out_W    = (const float*)d_in[15];
  const float* out_b    = (const float*)d_in[16];
  float* out = (float*)d_out;

  char* ws = (char*)d_ws;
  size_t off = 0;
  auto alloc = [&](size_t bytes) {
    char* p = ws + off;
    off = (off + bytes + 255) & ~(size_t)255;
    return p;
  };
  f16*   Cbuf   = (f16*)  alloc((size_t)N_EDGES * 128 * 2);
  float* xf     = (float*)alloc((size_t)N_ATOMS * AFEA * 4);
  bf16*  xbf    = (bf16*) alloc((size_t)N_ATOMS * AFEA * 2);
  float* summed = (float*)alloc((size_t)N_ATOMS * AFEA * 4);
  bf16*  abf    = (bf16*) alloc((size_t)N_ATOMS * 96 * 2);
  v16bf* wfe    = (v16bf*)alloc((size_t)3 * 4 * 32 * 32);
  v16bf* wfc    = (v16bf*)alloc((size_t)NCONV * 6 * 8 * 32 * 32);
  float* sumbuf = (float*)alloc(64 * 128 * 4);
  float* sqbuf  = (float*)alloc(64 * 128 * 4);
  float* s2sum  = (float*)alloc(64 * 4);
  float* s2sq   = (float*)alloc(64 * 4);
  float* scale1 = (float*)alloc(128 * 4);
  float* shift1 = (float*)alloc(128 * 4);
  float* scale2 = (float*)alloc(64 * 4);
  float* shift2 = (float*)alloc(64 * 4);
  float* csum   = (float*)alloc((size_t)N_CRYS * AFEA * 4);
  float* counts = (float*)alloc((size_t)N_CRYS * 4);

  prep_atoms_kernel<<<(N_ATOMS * 96 + 255) / 256, 256, 0, stream>>>(atom_fea, abf);
  build_wfrag_kernel<<<6, 64, 0, stream>>>(emb_W, wfe, ORIG, AFEA, 3, 4);
  for (int i = 0; i < NCONV; ++i)
    build_wfrag_kernel<<<24, 64, 0, stream>>>(conv_W + (size_t)i * 169 * 128,
                                              wfc + (size_t)i * 1536, 169, 128, 6, 8);
  emb_gemm_kernel<<<N_ATOMS / 32, 256, 0, stream>>>(abf, wfe, emb_b, xf, xbf);

  for (int i = 0; i < NCONV; ++i) {
    hipMemsetAsync(sumbuf, 0, 64 * 128 * 4, stream);
    hipMemsetAsync(sqbuf, 0, 64 * 128 * 4, stream);
    conv_gemm_kernel<<<N_EDGES / 64, 256, 0, stream>>>(
        xbf, nbr_fea, self_idx, nbr_idx, wfc + (size_t)i * 1536,
        conv_b + (size_t)i * 128, Cbuf, sumbuf, sqbuf);
    bn_finalize_kernel<<<1, 128, 0, stream>>>(sumbuf, sqbuf, 64, 128, 1.0f / N_EDGES,
                                              bn1_g + i * 128, bn1_b + i * 128, scale1, shift1);
    hipMemsetAsync(summed, 0, (size_t)N_ATOMS * AFEA * 4, stream);
    hipMemsetAsync(s2sum, 0, 64 * 4, stream);
    hipMemsetAsync(s2sq, 0, 64 * 4, stream);
    conv_msg_kernel<<<N_EDGES / 64, 256, 0, stream>>>(Cbuf, scale1, shift1, self_idx, summed);
    colstats64_kernel<<<256, 256, 0, stream>>>(summed, N_ATOMS, s2sum, s2sq);
    bn_finalize_kernel<<<1, 64, 0, stream>>>(s2sum, s2sq, 1, 64, 1.0f / N_ATOMS,
                                             bn2_g + i * 64, bn2_b + i * 64, scale2, shift2);
    update_x_kernel<<<N_ATOMS * AFEA / 256, 256, 0, stream>>>(summed, scale2, shift2, xf, xbf);
  }

  hipMemsetAsync(csum, 0, (size_t)N_CRYS * AFEA * 4, stream);
  hipMemsetAsync(counts, 0, (size_t)N_CRYS * 4, stream);
  pool_kernel<<<N_ATOMS / 4, 256, 0, stream>>>(xf, cry_idx, csum, counts);
  head_kernel<<<N_CRYS, 128, 0, stream>>>(csum, counts, fc1_W, fc1_b, out_W, out_b, out);

  (void)in_sizes; (void)n_in; (void)out_size; (void)ws_size;
}
